// RNN_26938034880941
// MI455X (gfx1250) — matrix operand order repaired and checked by reading, not compiled, not run on hardware
//
#include <hip/hip_runtime.h>
#include <cstdint>
#include <cstddef>

// ---------------------------------------------------------------------------
// Fused vanilla-RNN forward for MI455X (gfx1250, wave32).
//   h_t = tanh(h_{t-1} W_hh^T + x_t W_ih^T + b_ih + b_hh);  out = h_S W_fc^T + b_fc
// One wave owns a 16-row batch tile; each timestep is ONE v_wmma_f32_16x16x32_f16:
//   A (constant, 16x32 f16): rows = hidden-out; K0..15 = W_hh, K16 = W_ih, K17 = bias
//   B (per step,  32x16 f16): cols = batch;     K0..15 = h_{t-1},  K16 = x_t,  K17 = 1
//   D (16x16 f32) -> tanh -> packed f16 -> reused as-is for B K0..15 at t+1.
// ---------------------------------------------------------------------------

typedef __attribute__((ext_vector_type(16))) _Float16 v16h;
typedef __attribute__((ext_vector_type(8)))  float    v8f;
typedef __attribute__((ext_vector_type(2)))  __fp16   v2fp;   // matches cvt_pkrtz return type

__device__ __forceinline__ uint32_t pk2(float a, float b) {
  union { v2fp h; uint32_t u; } cv;
  cv.h = __builtin_amdgcn_cvt_pkrtz(a, b);   // v_cvt_pk_rtz_f16_f32
  return cv.u;
}

#if __has_builtin(__builtin_amdgcn_tanhf)
__device__ __forceinline__ float fast_tanh(float x) { return __builtin_amdgcn_tanhf(x); } // v_tanh_f32
#else
__device__ __forceinline__ float fast_tanh(float x) { return tanhf(x); }
#endif

union HMat { v16h h; uint32_t u[8]; };

__global__ __launch_bounds__(256, 1)
void rnn_wmma_kernel(const float* __restrict__ x,
                     const float* __restrict__ Wih,
                     const float* __restrict__ Whh,
                     const float* __restrict__ bih,
                     const float* __restrict__ bhh,
                     const float* __restrict__ Wfc,
                     const float* __restrict__ bfc,
                     float* __restrict__ out,
                     int B, int S)
{
  const int lane      = (int)(threadIdx.x & 31u);
  const int waveInBlk = (int)(threadIdx.x >> 5u);
  const int tile      = (int)blockIdx.x * (int)(blockDim.x >> 5u) + waveInBlk;
  const int b_base    = tile * 16;
  if (b_base >= B) return;                 // wave-uniform: EXEC stays all-1s below

  const int  m  = lane & 15;               // A: row M; B/D: column N (batch)
  const bool hi = lane >= 16;

  // ---- Build constant A matrix (augmented weights), ISA 16-bit A 16x32 layout:
  //   lanes 0-15 : row M=lane,    u[0..3]=K0..7,   u[4..7]=K16..23
  //   lanes16-31 : row M=lane-16, u[0..3]=K8..15,  u[4..7]=K24..31
  HMat A;
  const int kbase = hi ? 8 : 0;
  #pragma unroll
  for (int j = 0; j < 4; ++j) {
    float w0 = Whh[m * 16 + kbase + 2 * j];
    float w1 = Whh[m * 16 + kbase + 2 * j + 1];
    A.u[j] = pk2(w0, w1);
  }
  const float biasm = bih[m] + bhh[m];
  A.u[4] = hi ? 0u : pk2(Wih[m], biasm);   // K16 = W_ih (I==1), K17 = fused bias
  A.u[5] = 0u; A.u[6] = 0u; A.u[7] = 0u;

  // FC weights for this lane's 8 hidden rows of D (M = j + 8*hi)
  float wfc[8];
  #pragma unroll
  for (int j = 0; j < 8; ++j) wfc[j] = Wfc[j + (hi ? 8 : 0)];

  // Each lane streams one contiguous x row (x is [B,S] since I==1)
  const float* xrow = x + (size_t)(b_base + m) * (size_t)S;

  const int permIdx = (lane | 16) << 2;    // lanes<16 fetch from lane+16; hi lanes self

  uint32_t pk[4] = {0u, 0u, 0u, 0u};       // packed-f16 state (this lane's 8 D rows); h_0 = 0
  float tv[8] = {0,0,0,0,0,0,0,0};

  for (int t = 0; t < S; ++t) {
    float xv = xrow[t];
    __builtin_prefetch(xrow + t + 64, 0, 3);   // global_prefetch on the x stream

    // rows M=8..15 of our batch column live in lane+16 -> cross-lane gather
    uint32_t o0 = (uint32_t)__builtin_amdgcn_ds_bpermute(permIdx, (int)pk[0]);
    uint32_t o1 = (uint32_t)__builtin_amdgcn_ds_bpermute(permIdx, (int)pk[1]);
    uint32_t o2 = (uint32_t)__builtin_amdgcn_ds_bpermute(permIdx, (int)pk[2]);
    uint32_t o3 = (uint32_t)__builtin_amdgcn_ds_bpermute(permIdx, (int)pk[3]);

    // ISA 16-bit B 32x16 layout: lanes 0-15 hold K=0..7,16..23 of column N=lane,
    // lanes 16-31 hold K=8..15,24..31 of column N=lane-16.
    HMat Bm;
    const uint32_t xpack = pk2(xv, 1.0f);  // K16 = x_t, K17 = 1.0
    Bm.u[0] = pk[0];
    Bm.u[1] = pk[1];
    Bm.u[2] = pk[2];
    Bm.u[3] = pk[3];
    Bm.u[4] = hi ? 0u    : xpack;
    Bm.u[5] = 0u;
    Bm.u[6] = 0u;
    Bm.u[7] = 0u;

    v8f cz = {};                           // C = 0; bias already folded into A x B
    v8f d = __builtin_amdgcn_wmma_f32_16x16x32_f16(
        /*neg_a=*/false, A.h, /*neg_b=*/false, Bm.h,
        /*c_mod=*/(short)0, cz, /*reuse_a=*/false, /*reuse_b=*/false);

    #pragma unroll
    for (int j = 0; j < 8; ++j) tv[j] = fast_tanh(d[j]);
    pk[0] = pk2(tv[0], tv[1]);
    pk[1] = pk2(tv[2], tv[3]);
    pk[2] = pk2(tv[4], tv[5]);
    pk[3] = pk2(tv[6], tv[7]);
  }

  // Final FC (O==1): dot over 16 hidden = own 8 + partner lane's 8
  float partial = 0.f;
  #pragma unroll
  for (int j = 0; j < 8; ++j) partial += tv[j] * wfc[j];
  union { float f; int i; } pc; pc.f = partial;
  int oth = __builtin_amdgcn_ds_bpermute((lane ^ 16) << 2, pc.i);
  union { int i; float f; } oc; oc.i = oth;
  float total = partial + oc.f;

  if (!hi) out[b_base + m] = total + bfc[0];
}

extern "C" void kernel_launch(void* const* d_in, const int* in_sizes, int n_in,
                              void* d_out, int out_size, void* d_ws, size_t ws_size,
                              hipStream_t stream) {
  (void)n_in; (void)d_ws; (void)ws_size;
  const float* x   = (const float*)d_in[0];   // [B,S,1]
  const float* Wih = (const float*)d_in[1];   // [16,1]
  const float* Whh = (const float*)d_in[2];   // [16,16]
  const float* bih = (const float*)d_in[3];   // [16]
  const float* bhh = (const float*)d_in[4];   // [16]
  const float* Wfc = (const float*)d_in[5];   // [1,16]
  const float* bfc = (const float*)d_in[6];   // [1]
  float* out = (float*)d_out;                 // [B,1]

  const int B = out_size;                     // O == 1
  const int S = (B > 0) ? (in_sizes[0] / B) : 0;  // I == 1

  const int tiles = (B + 15) / 16;            // one wave32 per 16-batch tile
  const int wavesPerBlock = 8;                // 256 threads = 8 waves per block
  const int blocks = (tiles + wavesPerBlock - 1) / wavesPerBlock;

  rnn_wmma_kernel<<<blocks, wavesPerBlock * 32, 0, stream>>>(
      x, Wih, Whh, bih, bhh, Wfc, bfc, out, B, S);
}
